// ResidualDenoiser_14628658610537
// MI455X (gfx1250) — matrix operand order repaired and checked by reading, not compiled, not run on hardware
//
#include <hip/hip_runtime.h>

typedef __attribute__((ext_vector_type(16))) __bf16 v16bf;
typedef __attribute__((ext_vector_type(8)))  float  v8f;

#define LDT 40          // LDS row stride in bf16 elems (80B: 16B-aligned, conflict-padded)
#define NSEG 10
#define BATCH 8192
#define ACT_LD 3328

#if defined(__gfx1250__) && __has_builtin(__builtin_amdgcn_global_load_async_to_lds_b128)
#define HAVE_ASYNC_LDS 1
#else
#define HAVE_ASYNC_LDS 0
#endif

#if HAVE_ASYNC_LDS
typedef int v4i __attribute__((vector_size(16)));
typedef __attribute__((address_space(1))) v4i gv4i;   // global 16B vector
typedef __attribute__((address_space(3))) v4i sv4i;   // LDS 16B vector
#endif

union Frag { v16bf v; uint4 q[2]; };

__device__ __forceinline__ void wait_async0() {
#if HAVE_ASYNC_LDS
#if __has_builtin(__builtin_amdgcn_s_wait_asynccnt)
  __builtin_amdgcn_s_wait_asynccnt(0);
#else
  asm volatile("s_wait_asynccnt 0x0" ::: "memory");
#endif
#endif
}

// ---------------------------------------------------------------- conversions
__global__ __launch_bounds__(256) void cvt_bf16_kernel(
    const float* __restrict__ src, __bf16* __restrict__ dst,
    int rows, int cols, int dstStride, int dstColOff)
{
  int n = rows * cols;
  for (int i = blockIdx.x * blockDim.x + threadIdx.x; i < n; i += gridDim.x * blockDim.x) {
    int r = i / cols, c = i % cols;
    dst[(size_t)r * dstStride + dstColOff + c] = (__bf16)src[i];
  }
}

// ---------------------------------------------------------------- staging
#if HAVE_ASYNC_LDS
// gfx1250 async copy: global -> LDS with no VGPR staging (ASYNCcnt tracked)
__device__ __forceinline__ void stage_async(const __bf16* __restrict__ A, int lda,
                                            const __bf16* __restrict__ W, int K,
                                            int m0, int n0, int k0, int tid,
                                            __bf16* la, __bf16* lb)
{
#pragma unroll
  for (int u = 0; u < 2; ++u) {
    int c = tid + u * 256;
    int row = c >> 2, cc = (c & 3) << 3;          // 128 rows x 4 chunks of 8 bf16
    __bf16* ga = const_cast<__bf16*>(A + (size_t)(m0 + row) * lda + k0 + cc);
    __bf16* gb = const_cast<__bf16*>(W + (size_t)(n0 + row) * (size_t)K + k0 + cc);
    __builtin_amdgcn_global_load_async_to_lds_b128(
        (gv4i*)ga, (sv4i*)(la + row * LDT + cc), 0, 0);
    __builtin_amdgcn_global_load_async_to_lds_b128(
        (gv4i*)gb, (sv4i*)(lb + row * LDT + cc), 0, 0);
  }
}
#else
__device__ __forceinline__ void stage_load(const __bf16* __restrict__ A, int lda,
                                           const __bf16* __restrict__ W, int K,
                                           int m0, int n0, int k0, int tid,
                                           uint4* ra, uint4* rb)
{
#pragma unroll
  for (int u = 0; u < 2; ++u) {
    int c = tid + u * 256;
    int row = c >> 2, cc = (c & 3) << 3;
    ra[u] = *(const uint4*)(A + (size_t)(m0 + row) * lda + k0 + cc);
    rb[u] = *(const uint4*)(W + (size_t)(n0 + row) * (size_t)K + k0 + cc);
  }
}
__device__ __forceinline__ void stage_store(__bf16* la, __bf16* lb, int tid,
                                            const uint4* ra, const uint4* rb)
{
#pragma unroll
  for (int u = 0; u < 2; ++u) {
    int c = tid + u * 256;
    int row = c >> 2, cc = (c & 3) << 3;
    *(uint4*)(la + row * LDT + cc) = ra[u];
    *(uint4*)(lb + row * LDT + cc) = rb[u];
  }
}
#endif

// ---------------------------------------------------------------- fused GEMM
// C[M=8192, N] = A[M,K] * W[N,K]^T  (+ fused BN/ReLU -> bf16, or +bias -> f32)
__global__ __launch_bounds__(256) void wmma_mlp_kernel(
    const __bf16* __restrict__ A, int lda, int K,
    const __bf16* __restrict__ W, int nTilesN,
    const float* __restrict__ bias,
    const float* __restrict__ g,  const float* __restrict__ be,
    const float* __restrict__ rm, const float* __restrict__ rv,
    __bf16* __restrict__ outB, float* __restrict__ outF, int ldo)
{
  __shared__ __bf16 ldsA[2][128 * LDT];
  __shared__ __bf16 ldsB[2][128 * LDT];

  const int tid  = threadIdx.x;
  const int lane = tid & 31;
  const int wv   = tid >> 5;                       // 8 waves
  const int wm   = wv >> 2;                        // 0..1  -> 64 rows each
  const int wn   = wv & 3;                         // 0..3  -> 32 cols each
  const int half = lane >> 4;
  const int l15  = lane & 15;

  const int tile_n = blockIdx.x % nTilesN;
  const int tile_m = blockIdx.x / nTilesN;
  const int m0 = tile_m * 128;
  const int n0 = tile_n * 128;
  const int kTiles = K >> 5;

  v8f acc[4][2] = {};

#if HAVE_ASYNC_LDS
  stage_async(A, lda, W, K, m0, n0, 0, tid, ldsA[0], ldsB[0]);
  wait_async0();
  __syncthreads();
#else
  uint4 ra[2], rb[2];
  stage_load(A, lda, W, K, m0, n0, 0, tid, ra, rb);
  stage_store(ldsA[0], ldsB[0], tid, ra, rb);
  __syncthreads();
#endif

  for (int kt = 0; kt < kTiles; ++kt) {
    const int cur = kt & 1;
    const bool hasNext = (kt + 1) < kTiles;

#if HAVE_ASYNC_LDS
    if (hasNext)
      stage_async(A, lda, W, K, m0, n0, (kt + 1) << 5, tid, ldsA[cur ^ 1], ldsB[cur ^ 1]);
#else
    if (hasNext) stage_load(A, lda, W, K, m0, n0, (kt + 1) << 5, tid, ra, rb);
#endif

    Frag fa[4], fb[2];
#pragma unroll
    for (int i = 0; i < 4; ++i) {
      // A 16x32 bf16 frag: lane=(half,m); dwords half*4..+3 and 8+half*4..+3 of row
      const __bf16* rp = &ldsA[cur][(wm * 64 + i * 16 + l15) * LDT + half * 8];
      fa[i].q[0] = *(const uint4*)(rp);
      fa[i].q[1] = *(const uint4*)(rp + 16);
    }
#pragma unroll
    for (int j = 0; j < 2; ++j) {
      // B 32x16 bf16 frag: lane=(half,n); K elems half*8..+7 and 16+half*8..+7 of W row n
      const __bf16* bp = &ldsB[cur][(wn * 32 + j * 16 + l15) * LDT + half * 8];
      fb[j].q[0] = *(const uint4*)(bp);
      fb[j].q[1] = *(const uint4*)(bp + 16);
    }
#pragma unroll
    for (int i = 0; i < 4; ++i)
#pragma unroll
      for (int j = 0; j < 2; ++j)
        acc[i][j] = __builtin_amdgcn_wmma_f32_16x16x32_bf16(
            false, fa[i].v, false, fb[j].v, (short)0, acc[i][j], false, false);

#if HAVE_ASYNC_LDS
    if (hasNext) wait_async0();
#else
    if (hasNext) stage_store(ldsA[cur ^ 1], ldsB[cur ^ 1], tid, ra, rb);
#endif
    __syncthreads();
  }

  // -------- epilogue; C frag: lane=(half,n), vgpr r -> m_local = r + 8*half
  const int mB = m0 + wm * 64;
  const int nB = n0 + wn * 32;
  if (outB) {
#pragma unroll
    for (int j = 0; j < 2; ++j) {
      int n = nB + j * 16 + l15;
      float s = g[n] * rsqrtf(rv[n] + 1e-5f);
      float t = (bias[n] - rm[n]) * s + be[n];
#pragma unroll
      for (int i = 0; i < 4; ++i)
#pragma unroll
        for (int r = 0; r < 8; ++r) {
          int m = mB + i * 16 + half * 8 + r;
          float y = acc[i][j][r] * s + t;
          y = y > 0.f ? y : 0.f;
          outB[(size_t)m * ldo + n] = (__bf16)y;
        }
    }
  } else {
#pragma unroll
    for (int j = 0; j < 2; ++j) {
      int n = nB + j * 16 + l15;
      float bb = bias[n];
#pragma unroll
      for (int i = 0; i < 4; ++i)
#pragma unroll
        for (int r = 0; r < 8; ++r) {
          int m = mB + i * 16 + half * 8 + r;
          outF[(size_t)m * ldo + n] = acc[i][j][r] + bb;
        }
    }
  }
}

// ---------------------------------------------------------------- segmented softmax
__global__ __launch_bounds__(256) void segsoftmax_kernel(
    const float* __restrict__ h, const int* __restrict__ seg, float* __restrict__ out)
{
  __shared__ unsigned smax[NSEG];
  __shared__ float    ssum[NSEG];
  const int row = blockIdx.x, t = threadIdx.x;
  if (t < NSEG) { smax[t] = 0u; ssum[t] = 0.f; }
  __syncthreads();

  float v = h[(size_t)row * 256 + t];
  int   s = seg[t];
  unsigned e = __float_as_uint(v);
  e = (e & 0x80000000u) ? ~e : (e | 0x80000000u);   // order-preserving encode
  atomicMax(&smax[s], e);
  __syncthreads();

  unsigned me = smax[s];
  float m = __uint_as_float((me & 0x80000000u) ? (me & 0x7fffffffu) : ~me);
  float ex = __expf(v - m);
  atomicAdd(&ssum[s], ex);
  __syncthreads();

  out[(size_t)row * 256 + t] = ex / ssum[s];
}

// ---------------------------------------------------------------- launch
extern "C" void kernel_launch(void* const* d_in, const int* in_sizes, int n_in,
                              void* d_out, int out_size, void* d_ws, size_t ws_size,
                              hipStream_t stream) {
  const float* x   = (const float*)d_in[0];
  const float* W0  = (const float*)d_in[1];
  const float* b0  = (const float*)d_in[2];
  const float* g0  = (const float*)d_in[3];
  const float* be0 = (const float*)d_in[4];
  const float* rm0 = (const float*)d_in[5];
  const float* rv0 = (const float*)d_in[6];
  const float* W1  = (const float*)d_in[7];
  const float* b1  = (const float*)d_in[8];
  const float* g1  = (const float*)d_in[9];
  const float* be1 = (const float*)d_in[10];
  const float* rm1 = (const float*)d_in[11];
  const float* rv1 = (const float*)d_in[12];
  const float* W2  = (const float*)d_in[13];
  const float* b2  = (const float*)d_in[14];
  const float* g2  = (const float*)d_in[15];
  const float* be2 = (const float*)d_in[16];
  const float* rm2 = (const float*)d_in[17];
  const float* rv2 = (const float*)d_in[18];
  const float* W3  = (const float*)d_in[19];
  const float* b3  = (const float*)d_in[20];
  const int*   seg = (const int*)d_in[21];

  // workspace layout: Act[8192][3328] bf16 = [a2 | a1 | a0 | x], then bf16 weights, then h3 f32
  char* ws = (char*)d_ws;
  __bf16* Act = (__bf16*)ws;
  size_t off = (size_t)BATCH * ACT_LD * 2;
  __bf16* wB0 = (__bf16*)(ws + off); off += (size_t)1024 * 256  * 2;
  __bf16* wB1 = (__bf16*)(ws + off); off += (size_t)1024 * 1280 * 2;
  __bf16* wB2 = (__bf16*)(ws + off); off += (size_t)1024 * 2304 * 2;
  __bf16* wB3 = (__bf16*)(ws + off); off += (size_t)256  * 3328 * 2;
  float*  h3  = (float*)(ws + off);

  // f32 -> bf16 conversions
  cvt_bf16_kernel<<<2048, 256, 0, stream>>>(x,  Act, BATCH, 256, ACT_LD, 3072);
  cvt_bf16_kernel<<<1024, 256, 0, stream>>>(W0, wB0, 1024, 256,  256,  0);
  cvt_bf16_kernel<<<2048, 256, 0, stream>>>(W1, wB1, 1024, 1280, 1280, 0);
  cvt_bf16_kernel<<<2048, 256, 0, stream>>>(W2, wB2, 1024, 2304, 2304, 0);
  cvt_bf16_kernel<<<2048, 256, 0, stream>>>(W3, wB3, 256,  3328, 3328, 0);

  // layer 0: K=256  (reads x block, writes a0 block)
  wmma_mlp_kernel<<<512, 256, 0, stream>>>(Act + 3072, ACT_LD, 256,  wB0, 8,
      b0, g0, be0, rm0, rv0, Act + 2048, nullptr, ACT_LD);
  // layer 1: K=1280 (reads [a0|x], writes a1)
  wmma_mlp_kernel<<<512, 256, 0, stream>>>(Act + 2048, ACT_LD, 1280, wB1, 8,
      b1, g1, be1, rm1, rv1, Act + 1024, nullptr, ACT_LD);
  // layer 2: K=2304 (reads [a1|a0|x], writes a2)
  wmma_mlp_kernel<<<512, 256, 0, stream>>>(Act + 1024, ACT_LD, 2304, wB2, 8,
      b2, g2, be2, rm2, rv2, Act, nullptr, ACT_LD);
  // layer 3: K=3328, N=256, +bias -> f32
  wmma_mlp_kernel<<<128, 256, 0, stream>>>(Act, ACT_LD, 3328, wB3, 2,
      b3, nullptr, nullptr, nullptr, nullptr, nullptr, h3, 256);

  segsoftmax_kernel<<<BATCH, 256, 0, stream>>>(h3, seg, (float*)d_out);
}
